// DABlock_36481452212798
// MI455X (gfx1250) — matrix operand order repaired and checked by reading, not compiled, not run on hardware
//
#include <hip/hip_runtime.h>
#include <hip/hip_bf16.h>

// ============================================================================
// DABlock for MI455X (gfx1250, wave32, WMMA).
// All GEMM-shaped work (QKV, DTA clustering GEMMs, attention, proj, MLP, and
// both 3x3 convs as implicit GEMM) runs on v_wmma_f32_16x16x32_f16 with f32
// accumulate. GEMM A-tiles are staged into LDS by the Tensor Data Mover
// (tensor_load_to_lds + s_wait_tensorcnt) with TDM row padding reproducing the
// bank-conflict-avoiding 40-half LDS row stride; B-tiles are staged transposed
// with register-batched gathers (loads clause together, then LDS stores).
// Scratch requirement: ~103 MB carved from d_ws.
// ============================================================================

typedef __attribute__((ext_vector_type(16))) _Float16 v16h;
typedef __attribute__((ext_vector_type(8)))  float    v8f;
typedef __attribute__((ext_vector_type(4)))  unsigned int v4u;
typedef __attribute__((ext_vector_type(4)))  int      v4i;
typedef __attribute__((ext_vector_type(8)))  int      v8i;
typedef _Float16 h16;

#define CB   256      // C
#define NPIX 4096     // H*W
#define C4B  1024     // 4*C
#define KP   128      // KPOOL
#define NSTG 6        // STAGES
#define NHEAD 8

#if defined(__has_builtin)
#  if __has_builtin(__builtin_amdgcn_tensor_load_to_lds) && \
      __has_builtin(__builtin_amdgcn_s_wait_tensorcnt)
#    define HAVE_TDM 1
#  endif
#endif
#ifndef HAVE_TDM
#  define HAVE_TDM 0
#endif

__device__ __forceinline__ v8f wmma_f16(v16h a, v16h b, v8f c) {
  return __builtin_amdgcn_wmma_f32_16x16x32_f16(
      /*neg_a=*/false, a, /*neg_b=*/false, b,
      /*c_mod=*/(short)0, c, /*reuse_a=*/false, /*reuse_b=*/false);
}

// A fragment (16x32 f16, ISA 7.12.2): lane row = l%16, kbase=(l/16)*8,
// halves 0..7 -> k=kbase+j ; halves 8..15 -> k=16+kbase+(j-8).
// LDS tile is [row][40 halves] (80B row stride keeps 16B alignment, spreads banks).
__device__ __forceinline__ v16h load_frag_a(const h16* As, int row0) {
  const int lane = threadIdx.x & 31;
  const h16* p = As + (size_t)(row0 + (lane & 15)) * 40 + ((lane >> 4) << 3);
  union { v16h v; uint4 q[2]; } u;
  u.q[0] = *(const uint4*)(p);
  u.q[1] = *(const uint4*)(p + 16);
  return u.v;
}

// B fragment (32x16 f16): lane col = l%16, kbase=(l/16)*8, halves 0..7 -> k=kbase+j,
// halves 8..15 -> k=16+kbase+(j-8), from the transposed LDS tile Bs[col][40 halves].
__device__ __forceinline__ v16h load_frag_b(const h16* Bs, int col0) {
  const int lane = threadIdx.x & 31;
  const h16* p = Bs + (size_t)(col0 + (lane & 15)) * 40 + ((lane >> 4) << 3);
  union { v16h v; uint4 q[2]; } u;
  u.q[0] = *(const uint4*)(p);
  u.q[1] = *(const uint4*)(p + 16);
  return u.v;
}

#if HAVE_TDM
// Tensor Data Mover load of a 128-row x 32-half A tile into LDS, with TDM row
// padding (16 data DWORDs + 4 pad DWORDs per row -> 40-half LDS row stride).
// D# packing per CDNA5 ISA sec. 8.3/8.4; groups 2/3 are NULL-equivalent zeros
// (2D tensor). Issued once per workgroup (wave-uniform guard at call site).
__device__ __forceinline__ void tdm_load_tile_a(const h16* gsrc,
                                                unsigned ldsOff, int lda) {
  const unsigned long long ga = (unsigned long long)(uintptr_t)gsrc;
  v4u g0;
  g0.x = 1u;                                   // count=1, user descriptor
  g0.y = ldsOff;                               // lds_addr (bytes)
  g0.z = (unsigned)ga;                         // global_addr[31:0]
  g0.w = ((unsigned)(ga >> 32) & 0x01FFFFFFu)  // global_addr[56:32]
         | (2u << 30);                         // type = 2 ("image")
  const v8i g1 = {
    (int)((1u << 16)      // data_size = 2 bytes
        | (1u << 20)      // pad_enable
        | (3u << 22)      // pad_interval: 16 DWORDs of data
        | (3u << 25)),    // pad_amount:   4 DWORDs of padding
    0,                                // tensor_dim0[15:0] (=0 of 0x40000000)
    (int)0x4000u,                     // tensor_dim0[31:16] | tensor_dim1[15:0]
    (int)(0x4000u | (32u << 16)),     // tensor_dim1[31:16] | tile_dim0 = 32
    (int)128u,                        // tile_dim1 = 128 rows, tile_dim2 = 0
    lda,                              // tensor_dim0_stride[31:0] (elements)
    0, 0
  };
  const v4i z4 = {0, 0, 0, 0};
#if defined(__clang_major__) && __clang_major__ >= 23
  const v8i z8 = {0, 0, 0, 0, 0, 0, 0, 0};
  __builtin_amdgcn_tensor_load_to_lds(g0, g1, z4, z4, z8, 0);
#else
  __builtin_amdgcn_tensor_load_to_lds(g0, g1, z4, z4, 0);
#endif
}
#endif

// ============================================================================
// Generic z-batched WMMA GEMM: D = epilogue(alpha * A[MxK] @ B[KxN]).
// A,B f16 row-major (lda/ldb in elements). Block tile 128x64, K-step 32.
// 8 waves arranged 4x2, each wave computes a 32x32 quad of 16x16 WMMA tiles.
// Epilogue: v=acc*alpha; v=v*colscale+colbias; act(0 none,1 relu,2 erf-gelu);
//           v*=post_scale; v+=resid; store f32 and/or f16.
// All shapes used here divide BM/BN/BK exactly.
// ============================================================================
__global__ __launch_bounds__(256) void gemm_wmma(
    const h16* __restrict__ A, int lda, long long sA,
    const h16* __restrict__ B, int ldb, long long sB,
    float* __restrict__ Cf, h16* __restrict__ Ch, int ldc, long long sC,
    const float* __restrict__ colscale, const float* __restrict__ colbias,
    const float* __restrict__ resid, int ldr,
    int M, int N, int K, float alpha, float post_scale, int act)
{
  const int t  = threadIdx.x;
  const int z  = blockIdx.z;
  A += (size_t)z * sA;
  B += (size_t)z * sB;
  const int m0 = blockIdx.y * 128;
  const int n0 = blockIdx.x * 64;

  __shared__ __align__(16) h16 As[128 * 40];
  __shared__ __align__(16) h16 Bs[64 * 40];

  const int wid   = t >> 5;
  const int rbase = (wid >> 1) * 32;   // wave M offset within tile
  const int cbase = (wid & 1) * 32;    // wave N offset within tile

  v8f acc[2][2] = {};

  const int bcc = t & 63;              // B-tile column
  const int bk0 = t >> 6;
  // strength-reduced B gather pointer: walks 32*ldb per K-step, 4*ldb per row
  const h16* Bp = B + (size_t)bk0 * ldb + n0 + bcc;
#if HAVE_TDM
  const unsigned AsLdsOff = (unsigned)(uintptr_t)&As[0];  // low 32b = LDS offset
#else
  const int ar  = t >> 1;              // A-tile row this thread loads
  const int ach = (t & 1) * 16;        // half-chunk within the 32-wide K slice
#endif

  for (int k0 = 0; k0 < K; k0 += 32) {
#if HAVE_TDM
    // ---- async A tile via Tensor Data Mover (one issue per workgroup) ----
    if (wid == 0)
      tdm_load_tile_a(A + (size_t)m0 * lda + k0, AsLdsOff, lda);
#else
    // ---- manual A tile staging (128x32 halves), 32B per thread ----
    {
      const uint4* g = (const uint4*)(A + (size_t)(m0 + ar) * lda + k0 + ach);
      *(uint4*)&As[ar * 40 + ach]     = g[0];
      *(uint4*)&As[ar * 40 + ach + 8] = g[1];
    }
#endif
    // ---- stage B tile transposed: Bs[col][k] ----
    // Gather into registers first so the 8 global loads clause/pipeline,
    // then scatter to LDS (avoids load->ds_store->full-wait serialization).
    {
      const h16* bp = Bp;
      h16 tmp[8];
      #pragma unroll
      for (int i = 0; i < 8; ++i) { tmp[i] = *bp; bp += (size_t)4 * ldb; }
      #pragma unroll
      for (int i = 0; i < 8; ++i) Bs[bcc * 40 + bk0 + i * 4] = tmp[i];
      Bp += (size_t)32 * ldb;
    }
    if (k0 + 32 < K)   // gfx1250 global_prefetch_b8 of next B K-rows
      __builtin_prefetch((const void*)Bp, 0, 1);
#if HAVE_TDM
    if (wid == 0)
      __builtin_amdgcn_s_wait_tensorcnt(0);
#endif
    __syncthreads();

    const v16h a0 = load_frag_a(As, rbase);
    const v16h a1 = load_frag_a(As, rbase + 16);
    const v16h b0 = load_frag_b(Bs, cbase);
    const v16h b1 = load_frag_b(Bs, cbase + 16);
    acc[0][0] = wmma_f16(a0, b0, acc[0][0]);
    acc[0][1] = wmma_f16(a0, b1, acc[0][1]);
    acc[1][0] = wmma_f16(a1, b0, acc[1][0]);
    acc[1][1] = wmma_f16(a1, b1, acc[1][1]);
    __syncthreads();
  }

  // ---- epilogue (C layout: VGPR r -> row = r + 8*(lane>=16), col = lane%16) ----
  const int lane = t & 31;
  #pragma unroll
  for (int tm = 0; tm < 2; ++tm) {
    #pragma unroll
    for (int tn = 0; tn < 2; ++tn) {
      const int col = n0 + cbase + tn * 16 + (lane & 15);
      const float cs = colscale ? colscale[col] : 1.0f;
      const float cb = colbias  ? colbias[col]  : 0.0f;
      #pragma unroll
      for (int r = 0; r < 8; ++r) {
        const int row = m0 + rbase + tm * 16 + ((lane >> 4) << 3) + r;
        float v = acc[tm][tn][r] * alpha;
        v = v * cs + cb;
        if (act == 1)      v = fmaxf(v, 0.0f);
        else if (act == 2) v = 0.5f * v * (1.0f + erff(v * 0.70710678118654752f));
        v *= post_scale;
        if (resid) v += resid[(size_t)row * ldr + col];
        if (row < M && col < N) {
          const size_t off = (size_t)z * sC + (size_t)row * ldc + col;
          if (Cf) Cf[off] = v;
          if (Ch) Ch[off] = (h16)v;
        }
      }
    }
  }
}

// ============================================================================
// 3x3 conv (pad 1) as implicit GEMM: M = pixels (4096), N = cout (256),
// K = 9 * 256 cin. Input f16 channel-major [256][4096], weights pre-permuted
// to [kk][cin][cout] f16. Fused BN (scale/shift) + ReLU. Channel-major output.
// A staging stays manual (zero-padded im2col gather, not expressible as TDM),
// but gathers are register-batched so the loads clause together.
// ============================================================================
__global__ __launch_bounds__(256) void conv3x3_wmma(
    const h16* __restrict__ X, const h16* __restrict__ Wt,
    const float* __restrict__ bns, const float* __restrict__ bnt,
    h16* __restrict__ Yh, float* __restrict__ Yf)
{
  const int t  = threadIdx.x;
  const int m0 = blockIdx.y * 128;   // pixel tile
  const int n0 = blockIdx.x * 64;    // cout tile

  __shared__ __align__(16) h16 As[128 * 40];
  __shared__ __align__(16) h16 Bs[64 * 40];

  const int wid   = t >> 5;
  const int rbase = (wid >> 1) * 32;
  const int cbase = (wid & 1) * 32;

  v8f acc[2][2] = {};

  const int aci = t >> 3;          // cin lane within 32-wide K slice
  const int arb = (t & 7) * 16;    // 16-pixel block
  const int bcc = t & 63;
  const int bk0 = t >> 6;

  for (int kk = 0; kk < 9; ++kk) {
    const int ky = kk / 3 - 1, kx = kk % 3 - 1;
    for (int ci0 = 0; ci0 < 256; ci0 += 32) {
      // im2col gather with zero padding; loads first, LDS stores after
      {
        const h16* xp = X + (size_t)(ci0 + aci) * NPIX;
        h16 tmp[16];
        #pragma unroll
        for (int i = 0; i < 16; ++i) {
          const int p = m0 + arb + i;
          const int py = (p >> 6) + ky, px = (p & 63) + kx;
          h16 v = (h16)0.0f;
          if ((unsigned)py < 64u && (unsigned)px < 64u)
            v = xp[(py << 6) + px];
          tmp[i] = v;
        }
        #pragma unroll
        for (int i = 0; i < 16; ++i)
          As[(arb + i) * 40 + aci] = tmp[i];
      }
      // weight tile, register-batched like the GEMM B staging
      {
        const h16* wp = Wt + ((size_t)kk * 256 + ci0 + bk0) * 256 + n0 + bcc;
        h16 tmp[8];
        #pragma unroll
        for (int i = 0; i < 8; ++i) { tmp[i] = *wp; wp += (size_t)4 * 256; }
        #pragma unroll
        for (int i = 0; i < 8; ++i) Bs[bcc * 40 + bk0 + i * 4] = tmp[i];
      }
      __syncthreads();
      const v16h a0 = load_frag_a(As, rbase);
      const v16h a1 = load_frag_a(As, rbase + 16);
      const v16h b0 = load_frag_b(Bs, cbase);
      const v16h b1 = load_frag_b(Bs, cbase + 16);
      acc[0][0] = wmma_f16(a0, b0, acc[0][0]);
      acc[0][1] = wmma_f16(a0, b1, acc[0][1]);
      acc[1][0] = wmma_f16(a1, b0, acc[1][0]);
      acc[1][1] = wmma_f16(a1, b1, acc[1][1]);
      __syncthreads();
    }
  }

  const int lane = t & 31;
  #pragma unroll
  for (int tm = 0; tm < 2; ++tm) {
    #pragma unroll
    for (int tn = 0; tn < 2; ++tn) {
      const int col = n0 + cbase + tn * 16 + (lane & 15);
      const float s  = bns[col], sh = bnt[col];
      #pragma unroll
      for (int r = 0; r < 8; ++r) {
        const int row = m0 + rbase + tm * 16 + ((lane >> 4) << 3) + r;
        float v = fmaxf(acc[tm][tn][r] * s + sh, 0.0f);
        if (Yh) Yh[(size_t)col * NPIX + row] = (h16)v;
        if (Yf) Yf[(size_t)col * NPIX + row] = v;
      }
    }
  }
}

// ============================================================================
// Small VALU helper kernels
// ============================================================================

// LayerNorm over C=256; one token per 256-thread block.
// transposed!=0 : in[c*ldT + p] (reads NCHW image); else in[p*256 + c].
__global__ void ln_kernel(const float* __restrict__ in, int transposed, int ldT,
                          const float* __restrict__ g, const float* __restrict__ b,
                          h16* __restrict__ outh)
{
  const int p = blockIdx.x, c = threadIdx.x;
  const float v = transposed ? in[(size_t)c * ldT + p] : in[(size_t)p * CB + c];
  float s = v, s2 = v * v;
  #pragma unroll
  for (int o = 16; o > 0; o >>= 1) { s += __shfl_xor(s, o); s2 += __shfl_xor(s2, o); }
  __shared__ float rs[8], rs2[8];
  const int w = threadIdx.x >> 5, lane = threadIdx.x & 31;
  if (lane == 0) { rs[w] = s; rs2[w] = s2; }
  __syncthreads();
  float S = 0.f, S2 = 0.f;
  #pragma unroll
  for (int i = 0; i < 8; ++i) { S += rs[i]; S2 += rs2[i]; }
  const float mu  = S * (1.0f / CB);
  const float var = S2 * (1.0f / CB) - mu * mu;
  const float y = (v - mu) * rsqrtf(var + 1e-5f) * g[c] + b[c];
  outh[(size_t)p * CB + c] = (h16)y;
}

// Row softmax over 128 columns; one wave per row (4 cols/lane), 8 rows/block.
__global__ void softmax128(const float* __restrict__ in,
                           float* __restrict__ outf, h16* __restrict__ outh, int R)
{
  const int w = threadIdx.x >> 5, lane = threadIdx.x & 31;
  const int row = blockIdx.x * 8 + w;
  if (row >= R) return;
  const float* p = in + (size_t)row * KP;
  float v[4], mx = -1e30f;
  #pragma unroll
  for (int j = 0; j < 4; ++j) { v[j] = p[lane + 32 * j]; mx = fmaxf(mx, v[j]); }
  #pragma unroll
  for (int o = 16; o > 0; o >>= 1) mx = fmaxf(mx, __shfl_xor(mx, o));
  float s = 0.f;
  #pragma unroll
  for (int j = 0; j < 4; ++j) { v[j] = __expf(v[j] - mx); s += v[j]; }
  #pragma unroll
  for (int o = 16; o > 0; o >>= 1) s += __shfl_xor(s, o);
  const float inv = 1.0f / s;
  #pragma unroll
  for (int j = 0; j < 4; ++j) {
    const float r = v[j] * inv;
    if (outf) outf[(size_t)row * KP + lane + 32 * j] = r;
    if (outh) outh[(size_t)row * KP + lane + 32 * j] = (h16)r;
  }
}

// Column L2-normalize (axis over M rows) of a [M x 128] f32 matrix -> f16.
// Matches reference _l2norm: x / (1e-6 + ||x||).
__global__ void colnorm128(const float* __restrict__ in, h16* __restrict__ out, int M)
{
  const int col = blockIdx.x;
  float s = 0.f;
  for (int r = threadIdx.x; r < M; r += 256) {
    const float v = in[(size_t)r * KP + col];
    s += v * v;
  }
  #pragma unroll
  for (int o = 16; o > 0; o >>= 1) s += __shfl_xor(s, o);
  __shared__ float red[8];
  const int w = threadIdx.x >> 5, lane = threadIdx.x & 31;
  if (lane == 0) red[w] = s;
  __syncthreads();
  float tot = 0.f;
  #pragma unroll
  for (int i = 0; i < 8; ++i) tot += red[i];
  const float inv = 1.0f / (1e-6f + sqrtf(tot));
  for (int r = threadIdx.x; r < M; r += 256)
    out[(size_t)r * KP + col] = (h16)(in[(size_t)r * KP + col] * inv);
}

// AdaptiveMaxPool1d N=4096 -> 128 (window 32), channel-major f16 in, f32 out.
__global__ void poolmax(const h16* __restrict__ in, float* __restrict__ out)
{
  const int idx = blockIdx.x * 256 + threadIdx.x;   // c*128 + j, c<1024
  const int c = idx >> 7, j = idx & 127;
  const h16* p = in + (size_t)c * NPIX + j * 32;
  float m = -1e30f;
  #pragma unroll
  for (int i = 0; i < 32; ++i) m = fmaxf(m, (float)p[i]);
  out[idx] = m;
}

// f16 tiled transpose: in [R x Cn] -> out [Cn x R]. grid (Cn/32, R/32), 256 thr.
__global__ void transpose_f16(const h16* __restrict__ in, h16* __restrict__ out,
                              int R, int Cn)
{
  __shared__ h16 tile[32][33];
  const int bx = blockIdx.x * 32, by = blockIdx.y * 32;
  const int tx = threadIdx.x & 31, ty = threadIdx.x >> 5;
  #pragma unroll
  for (int i = 0; i < 32; i += 8)
    tile[ty + i][tx] = in[(size_t)(by + ty + i) * Cn + bx + tx];
  __syncthreads();
  #pragma unroll
  for (int i = 0; i < 32; i += 8)
    out[(size_t)(bx + ty + i) * R + by + tx] = tile[tx][ty + i];
}

// f32 -> f16 transpose-convert (weight prep): in [R x Cn] f32 -> out [Cn x R] f16.
__global__ void tconv_f32h(const float* __restrict__ in, h16* __restrict__ out,
                           int R, int Cn)
{
  __shared__ float tile[32][33];
  const int bx = blockIdx.x * 32, by = blockIdx.y * 32;
  const int tx = threadIdx.x & 31, ty = threadIdx.x >> 5;
  #pragma unroll
  for (int i = 0; i < 32; i += 8)
    tile[ty + i][tx] = in[(size_t)(by + ty + i) * Cn + bx + tx];
  __syncthreads();
  #pragma unroll
  for (int i = 0; i < 32; i += 8)
    out[(size_t)(bx + ty + i) * R + by + tx] = (h16)tile[tx][ty + i];
}

// f32 -> f16 straight convert.
__global__ void conv_to_h(const float* __restrict__ in, h16* __restrict__ out, int n)
{
  const int i = blockIdx.x * 256 + threadIdx.x;
  if (i < n) out[i] = (h16)in[i];
}

// Conv weight permute: OIHW f32 -> [kk][cin][cout] f16.
__global__ void convw_prep(const float* __restrict__ w, h16* __restrict__ wt)
{
  const int idx = blockIdx.x * 256 + threadIdx.x;   // ((o*256+i)*9)+kk
  if (idx >= 256 * 256 * 9) return;
  const int o = idx / (256 * 9);
  const int rem = idx % (256 * 9);
  const int i = rem / 9, kk = rem % 9;
  wt[((size_t)kk * 256 + i) * 256 + o] = (h16)w[idx];
}

// Fold inference BN into scale/shift.
__global__ void bn_prep(const float* __restrict__ g, const float* __restrict__ b,
                        const float* __restrict__ m, const float* __restrict__ v,
                        float* __restrict__ s, float* __restrict__ sh)
{
  const int c = blockIdx.x * 256 + threadIdx.x;
  if (c < 256) {
    const float sc = g[c] * rsqrtf(v[c] + 1e-5f);
    s[c] = sc;
    sh[c] = b[c] - m[c] * sc;
  }
}

// out[c][p] = afin[p][c] + alpha * xc[c][p]
__global__ void final_combine(const float* __restrict__ afin,
                              const float* __restrict__ xc,
                              const float* __restrict__ alpha,
                              float* __restrict__ out)
{
  const int idx = blockIdx.x * 256 + threadIdx.x;   // c*4096 + p
  const int c = idx >> 12, p = idx & 4095;
  out[idx] = afin[(size_t)p * CB + c] + alpha[0] * xc[idx];
}

// ============================================================================
// Host orchestration
// ============================================================================
extern "C" void kernel_launch(void* const* d_in, const int* in_sizes, int n_in,
                              void* d_out, int out_size, void* d_ws, size_t ws_size,
                              hipStream_t stream)
{
  (void)in_sizes; (void)n_in; (void)out_size; (void)ws_size;

  const float* x   = (const float*)d_in[0];
  const float* Wq  = (const float*)d_in[1];
  const float* Wk  = (const float*)d_in[2];
  const float* Wv  = (const float*)d_in[3];
  const float* Wp  = (const float*)d_in[4];
  const float* bp  = (const float*)d_in[5];
  const float* g1  = (const float*)d_in[6];
  const float* b1  = (const float*)d_in[7];
  const float* g2  = (const float*)d_in[8];
  const float* b2  = (const float*)d_in[9];
  const float* W1  = (const float*)d_in[10];
  const float* bf1 = (const float*)d_in[11];
  const float* W2  = (const float*)d_in[12];
  const float* bf2 = (const float*)d_in[13];
  const float* cw1 = (const float*)d_in[14];
  const float* bg1 = (const float*)d_in[15];
  const float* bb1 = (const float*)d_in[16];
  const float* bm1 = (const float*)d_in[17];
  const float* bv1 = (const float*)d_in[18];
  const float* cw2 = (const float*)d_in[19];
  const float* bg2 = (const float*)d_in[20];
  const float* bb2 = (const float*)d_in[21];
  const float* bm2 = (const float*)d_in[22];
  const float* bv2 = (const float*)d_in[23];
  const float* alphap = (const float*)d_in[24];
  float* out = (float*)d_out;

  // ---- bump allocator over d_ws (needs ~103 MB) ----
  char* cur = (char*)d_ws;
  auto alloc = [&](size_t n) -> void* {
    uintptr_t p = ((uintptr_t)cur + 255) & ~(uintptr_t)255;
    cur = (char*)p + n;
    return (void*)p;
  };

  // persistent (weight) buffers
  h16* wqT  = (h16*)alloc((size_t)CB * C4B * 2);     // [256][1024]
  h16* wkT  = (h16*)alloc((size_t)CB * C4B * 2);
  h16* wvT  = (h16*)alloc((size_t)CB * C4B * 2);
  h16* wpT  = (h16*)alloc((size_t)C4B * CB * 2);     // [1024][256]
  h16* w1T  = (h16*)alloc((size_t)CB * C4B * 2);
  h16* w2T  = (h16*)alloc((size_t)C4B * CB * 2);
  h16* cwt1 = (h16*)alloc((size_t)9 * 256 * 256 * 2);
  h16* cwt2 = (h16*)alloc((size_t)9 * 256 * 256 * 2);
  float* bn1s = (float*)alloc(256 * 4);
  float* bn1t = (float*)alloc(256 * 4);
  float* bn2s = (float*)alloc(256 * 4);
  float* bn2t = (float*)alloc(256 * 4);

  // per-batch scratch (reused across the 4 batches)
  h16* xh    = (h16*)alloc((size_t)CB * NPIX * 2);
  h16* xc1h  = (h16*)alloc((size_t)CB * NPIX * 2);
  float* xcf = (float*)alloc((size_t)CB * NPIX * 4);
  h16* t1h   = (h16*)alloc((size_t)NPIX * CB * 2);
  h16* qnc   = (h16*)alloc((size_t)NPIX * C4B * 2);
  h16* knc   = (h16*)alloc((size_t)NPIX * C4B * 2);
  h16* vnc   = (h16*)alloc((size_t)NPIX * C4B * 2);
  h16* qcn   = (h16*)alloc((size_t)C4B * NPIX * 2);
  h16* kcn   = (h16*)alloc((size_t)C4B * NPIX * 2);
  h16* vcn   = (h16*)alloc((size_t)C4B * NPIX * 2);
  float* Sbuf = (float*)alloc((size_t)NPIX * KP * 4);
  h16* zh     = (h16*)alloc((size_t)NPIX * KP * 2);
  float* kbf  = (float*)alloc((size_t)C4B * KP * 4);
  h16* bas1h  = (h16*)alloc((size_t)C4B * KP * 2);
  h16* bas2h  = (h16*)alloc((size_t)C4B * KP * 2);
  h16* bas1T  = (h16*)alloc((size_t)KP * C4B * 2);
  float* attf = (float*)alloc((size_t)C4B * KP * 4);
  h16* atth   = (h16*)alloc((size_t)C4B * KP * 2);
  h16* ocn    = (h16*)alloc((size_t)C4B * NPIX * 2);
  h16* onc    = (h16*)alloc((size_t)NPIX * C4B * 2);
  float* a2   = (float*)alloc((size_t)NPIX * CB * 4);
  h16* ln2h   = (h16*)alloc((size_t)NPIX * CB * 2);
  h16* h1h    = (h16*)alloc((size_t)NPIX * C4B * 2);
  float* afin = (float*)alloc((size_t)NPIX * CB * 4);
  float* pbuf = (float*)alloc((size_t)C4B * KP * 4);

  const float SCALE = 0.17677669529663687f;   // (C/HEADS)^-0.5 = 1/sqrt(32)

  auto gemm = [&](const h16* A, int lda, long long sA,
                  const h16* B, int ldb, long long sB,
                  float* Cf, h16* Ch, int ldc, long long sC,
                  const float* cs, const float* cb, const float* rs, int ldr,
                  int M, int N, int K, float alpha, float ps, int act, int Z) {
    dim3 grid(N / 64, (M + 127) / 128, Z);
    gemm_wmma<<<grid, dim3(256), 0, stream>>>(A, lda, sA, B, ldb, sB,
                                              Cf, Ch, ldc, sC, cs, cb, rs, ldr,
                                              M, N, K, alpha, ps, act);
  };

  // ---------------- one-time prep ----------------
  tconv_f32h<<<dim3(CB / 32, C4B / 32), 256, 0, stream>>>(Wq, wqT, C4B, CB);
  tconv_f32h<<<dim3(CB / 32, C4B / 32), 256, 0, stream>>>(Wk, wkT, C4B, CB);
  tconv_f32h<<<dim3(CB / 32, C4B / 32), 256, 0, stream>>>(Wv, wvT, C4B, CB);
  tconv_f32h<<<dim3(C4B / 32, CB / 32), 256, 0, stream>>>(Wp, wpT, CB, C4B);
  tconv_f32h<<<dim3(CB / 32, C4B / 32), 256, 0, stream>>>(W1, w1T, C4B, CB);
  tconv_f32h<<<dim3(C4B / 32, CB / 32), 256, 0, stream>>>(W2, w2T, CB, C4B);
  convw_prep<<<(256 * 256 * 9 + 255) / 256, 256, 0, stream>>>(cw1, cwt1);
  convw_prep<<<(256 * 256 * 9 + 255) / 256, 256, 0, stream>>>(cw2, cwt2);
  bn_prep<<<1, 256, 0, stream>>>(bg1, bb1, bm1, bv1, bn1s, bn1t);
  bn_prep<<<1, 256, 0, stream>>>(bg2, bb2, bm2, bv2, bn2s, bn2t);

  // ---------------- per-batch pipeline ----------------
  for (int b = 0; b < 4; ++b) {
    const float* xb  = x + (size_t)b * CB * NPIX;
    float* outb      = out + (size_t)b * CB * NPIX;

    // conv residual branch (implicit-GEMM WMMA, fused BN+ReLU)
    conv_to_h<<<CB * NPIX / 256, 256, 0, stream>>>(xb, xh, CB * NPIX);
    conv3x3_wmma<<<dim3(4, 32), 256, 0, stream>>>(xh, cwt1, bn1s, bn1t, xc1h, nullptr);
    conv3x3_wmma<<<dim3(4, 32), 256, 0, stream>>>(xc1h, cwt2, bn2s, bn2t, nullptr, xcf);

    // t1 = LN(transpose(x)) -> f16
    ln_kernel<<<NPIX, 256, 0, stream>>>(xb, 1, NPIX, g1, b1, t1h);

    // QKV projections [4096x256]@[256x1024]
    gemm(t1h, CB, 0, wqT, C4B, 0, nullptr, qnc, C4B, 0,
         nullptr, nullptr, nullptr, 0, NPIX, C4B, CB, 1.f, 1.f, 0, 1);
    gemm(t1h, CB, 0, wkT, C4B, 0, nullptr, knc, C4B, 0,
         nullptr, nullptr, nullptr, 0, NPIX, C4B, CB, 1.f, 1.f, 0, 1);
    gemm(t1h, CB, 0, wvT, C4B, 0, nullptr, vnc, C4B, 0,
         nullptr, nullptr, nullptr, 0, NPIX, C4B, CB, 1.f, 1.f, 0, 1);

    // channel-major views
    transpose_f16<<<dim3(C4B / 32, NPIX / 32), 256, 0, stream>>>(qnc, qcn, NPIX, C4B);
    transpose_f16<<<dim3(C4B / 32, NPIX / 32), 256, 0, stream>>>(knc, kcn, NPIX, C4B);
    transpose_f16<<<dim3(C4B / 32, NPIX / 32), 256, 0, stream>>>(vnc, vcn, NPIX, C4B);

    // DTA init: bases = l2norm(maxpool(stream), axis=channels)
    poolmax<<<C4B * KP / 256, 256, 0, stream>>>(qcn, pbuf);
    colnorm128<<<KP, 256, 0, stream>>>(pbuf, bas1h, C4B);
    poolmax<<<C4B * KP / 256, 256, 0, stream>>>(kcn, pbuf);
    colnorm128<<<KP, 256, 0, stream>>>(pbuf, bas2h, C4B);

    // DTA EM iterations (the dominant GEMMs: 4096x1024x128 and 1024x128x4096)
    for (int s = 0; s < NSTG; ++s) {
      // stream 1 (x1 = q projection -> bas1)
      gemm(qnc, C4B, 0, bas1h, KP, 0, Sbuf, nullptr, KP, 0,
           nullptr, nullptr, nullptr, 0, NPIX, KP, C4B, 1.f, 1.f, 0, 1);
      softmax128<<<NPIX / 8, 256, 0, stream>>>(Sbuf, Sbuf, nullptr, NPIX);
      colnorm128<<<KP, 256, 0, stream>>>(Sbuf, zh, NPIX);
      gemm(qcn, NPIX, 0, zh, KP, 0, kbf, nullptr, KP, 0,
           nullptr, nullptr, nullptr, 0, C4B, KP, NPIX, 1.f, 1.f, 0, 1);
      colnorm128<<<KP, 256, 0, stream>>>(kbf, bas1h, C4B);
      // stream 2 (x2 = k projection -> bas2)
      gemm(knc, C4B, 0, bas2h, KP, 0, Sbuf, nullptr, KP, 0,
           nullptr, nullptr, nullptr, 0, NPIX, KP, C4B, 1.f, 1.f, 0, 1);
      softmax128<<<NPIX / 8, 256, 0, stream>>>(Sbuf, Sbuf, nullptr, NPIX);
      colnorm128<<<KP, 256, 0, stream>>>(Sbuf, zh, NPIX);
      gemm(kcn, NPIX, 0, zh, KP, 0, kbf, nullptr, KP, 0,
           nullptr, nullptr, nullptr, 0, C4B, KP, NPIX, 1.f, 1.f, 0, 1);
      colnorm128<<<KP, 256, 0, stream>>>(kbf, bas2h, C4B);
    }

    // att = softmax(qh @ kh^T * SCALE); qh=bas2 (from k-proj), kh=bas1 (from q-proj)
    transpose_f16<<<dim3(KP / 32, C4B / 32), 256, 0, stream>>>(bas1h, bas1T, C4B, KP);
    gemm(bas2h, KP, (long long)128 * 128, bas1T, C4B, 128,
         attf, nullptr, KP, (long long)128 * 128,
         nullptr, nullptr, nullptr, 0, 128, 128, 128, SCALE, 1.f, 0, NHEAD);
    softmax128<<<C4B / 8, 256, 0, stream>>>(attf, nullptr, atth, C4B);

    // o = att @ vh  (per-head, z-batched; channel-major output)
    gemm(atth, KP, (long long)128 * 128, vcn, NPIX, (long long)128 * NPIX,
         nullptr, ocn, NPIX, (long long)128 * NPIX,
         nullptr, nullptr, nullptr, 0, 128, NPIX, 128, 1.f, 1.f, 0, NHEAD);
    transpose_f16<<<dim3(NPIX / 32, C4B / 32), 256, 0, stream>>>(ocn, onc, C4B, NPIX);

    // a2 = 2 * relu(o @ Wp^T + bp)   (post_scale=2 folds the a=a+a)
    gemm(onc, C4B, 0, wpT, CB, 0, a2, nullptr, CB, 0,
         nullptr, bp, nullptr, 0, NPIX, CB, C4B, 1.f, 2.f, 1, 1);

    // MLP with pre-LN and residual
    ln_kernel<<<NPIX, 256, 0, stream>>>(a2, 0, 0, g2, b2, ln2h);
    gemm(ln2h, CB, 0, w1T, C4B, 0, nullptr, h1h, C4B, 0,
         nullptr, bf1, nullptr, 0, NPIX, C4B, CB, 1.f, 1.f, 2, 1);          // erf-GELU
    gemm(h1h, C4B, 0, w2T, CB, 0, afin, nullptr, CB, 0,
         nullptr, bf2, a2, CB, NPIX, CB, C4B, 1.f, 1.f, 0, 1);              // + residual

    // out = transpose(afin) + alpha * conv_branch
    final_combine<<<CB * NPIX / 256, 256, 0, stream>>>(afin, xcf, alphap, outb);
  }
}
